// GraphTransFormer_68676527063386
// MI455X (gfx1250) — matrix operand order repaired and checked by reading, not compiled, not run on hardware
//
#include <hip/hip_runtime.h>
#include <hip/hip_bf16.h>

typedef __attribute__((ext_vector_type(16))) _Float16 v16h;
typedef __attribute__((ext_vector_type(8)))  float    v8f;

#define N_NODES 8192
#define F_IN    512
#define DDIM    256
#define NCLS    40
#define LN_EPS  1e-5f

// ---------------------------------------------------------------------------
// WMMA helpers (layouts per CDNA5 ISA §7.12.2, wave32)
// ---------------------------------------------------------------------------
__device__ __forceinline__ v8f wmma_f16(v16h a, v16h b, v8f c) {
  return __builtin_amdgcn_wmma_f32_16x16x32_f16(
      /*neg_a=*/false, a, /*neg_b=*/false, b,
      /*c_mod=*/(short)0, c, /*reuse_a=*/false, /*reuse_b=*/false);
}

// A fragment: 16x32 f16, row-major source (ld in elements).
__device__ __forceinline__ v16h load_fragA(const _Float16* atile, int ld, int lane) {
  int r = lane & 15;
  int sel = lane >> 4;
  const _Float16* base = atile + r * ld + sel * 8;
  union { uint4 u[2]; v16h h; } f;
  f.u[0] = *reinterpret_cast<const uint4*>(base);
  f.u[1] = *reinterpret_cast<const uint4*>(base + 16);
  return f.h;
}

// B fragment: 32x16 f16 taken from B^T stored row-major [n][k] (ld = K stride).
__device__ __forceinline__ v16h load_fragB(const _Float16* bttile, int ld, int lane) {
  int n = lane & 15;
  int sel = lane >> 4;
  const uint4* p = reinterpret_cast<const uint4*>(bttile + (size_t)n * ld + sel * 8);
  union { uint4 u[2]; v16h h; } f;
  f.u[0] = p[0];
  f.u[1] = p[2];
  return f.h;
}

// Async global->LDS copy (CDNA5 ASYNCcnt path), 4 bytes per lane.
__device__ __forceinline__ void async_copy_b32(unsigned lds_addr, const void* gaddr) {
  asm volatile("global_load_async_to_lds_b32 %0, %1, off"
               :: "v"(lds_addr), "v"(gaddr)
               : "memory");
}
__device__ __forceinline__ void wait_asynccnt0() {
  asm volatile("s_wait_asynccnt 0x0" ::: "memory");
}

// ---------------------------------------------------------------------------
// Prep kernels
// ---------------------------------------------------------------------------
__global__ void cvt_f32_to_f16_kernel(const float* __restrict__ s,
                                      _Float16* __restrict__ d, int n) {
  int i = blockIdx.x * blockDim.x + threadIdx.x;
  if (i < n) d[i] = (_Float16)s[i];
}

// dst[c][r] = (f16) src[r][c]   (src is [R][C] row-major)
__global__ void transpose_f32_to_f16_kernel(const float* __restrict__ src,
                                            _Float16* __restrict__ dst,
                                            int R, int C) {
  int i = blockIdx.x * blockDim.x + threadIdx.x;
  if (i >= R * C) return;
  int r = i / C, c = i % C;
  dst[(size_t)c * R + r] = (_Float16)src[i];
}

// ---------------------------------------------------------------------------
// Generic GEMM: out[16 rows x 256 cols] = A[M,K](f16) @ Bt[256,K](f16)^T + bias
// 256 threads = 8 waves, wave w computes n-tiles 2w,2w+1.
// A tile staged into LDS via GLOBAL_LOAD_ASYNC_TO_LDS_B32 (ASYNCcnt).
// Optional fused LayerNorm (over 256 cols), f32 / f16 / f16-transposed stores.
// ---------------------------------------------------------------------------
template <bool DO_LN, bool ST_F32, bool ST_F16, bool ST_F16T>
__global__ void gemm_row16_kernel(const _Float16* __restrict__ A, int K,
                                  const _Float16* __restrict__ Bt,
                                  const float* __restrict__ bias,
                                  float* __restrict__ outF,
                                  _Float16* __restrict__ outH,
                                  _Float16* __restrict__ outHT, int Mtot) {
  __shared__ _Float16 As[16 * 32];
  __shared__ float Cs[16 * DDIM];

  const int row0 = blockIdx.x * 16;
  const int t = threadIdx.x;
  const int wave = t >> 5;
  const int lane = t & 31;
  const int n0 = wave * 32;

  v8f acc0 = {}, acc1 = {};

  // staging indices: each thread owns 2 consecutive halfs (4B) of the A tile
  const int sr = t >> 4;
  const int sk = (t & 15) * 2;
  const _Float16* asrc = A + (size_t)(row0 + sr) * K + sk;
  const unsigned my_lds = (unsigned)(uintptr_t)(&As[0]) + (unsigned)(sr * 32 + sk) * 2u;

  for (int k0 = 0; k0 < K; k0 += 32) {
    async_copy_b32(my_lds, asrc + k0);     // mem -> LDS, no VGPR round-trip
    wait_asynccnt0();
    __syncthreads();
    v16h a  = load_fragA(As, 32, lane);
    v16h b0 = load_fragB(Bt + (size_t)n0 * K + k0, K, lane);
    v16h b1 = load_fragB(Bt + (size_t)(n0 + 16) * K + k0, K, lane);
    acc0 = wmma_f16(a, b0, acc0);
    acc1 = wmma_f16(a, b1, acc1);
    __syncthreads();
  }

  // spill C tile (+bias) into LDS: element e -> row e+8*(lane>>4), col lane&15
  {
    int colw = lane & 15;
    int rh = (lane >> 4) * 8;
    float bb0 = bias[n0 + colw];
    float bb1 = bias[n0 + 16 + colw];
#pragma unroll
    for (int e = 0; e < 8; e++) {
      Cs[(e + rh) * DDIM + n0 + colw]      = acc0[e] + bb0;
      Cs[(e + rh) * DDIM + n0 + 16 + colw] = acc1[e] + bb1;
    }
  }
  __syncthreads();

  if constexpr (DO_LN) {
    // wave w normalizes rows 2w and 2w+1; lane covers 8 columns each
#pragma unroll
    for (int rr = 0; rr < 2; rr++) {
      int r = wave * 2 + rr;
      float xs[8];
      float s = 0.f, s2 = 0.f;
#pragma unroll
      for (int j = 0; j < 8; j++) {
        float x = Cs[r * DDIM + lane * 8 + j];
        xs[j] = x; s += x; s2 += x * x;
      }
#pragma unroll
      for (int off = 16; off; off >>= 1) {
        s  += __shfl_xor(s,  off, 32);
        s2 += __shfl_xor(s2, off, 32);
      }
      float mu  = s * (1.f / DDIM);
      float var = s2 * (1.f / DDIM) - mu * mu;
      float rs  = rsqrtf(var + LN_EPS);
#pragma unroll
      for (int j = 0; j < 8; j++)
        Cs[r * DDIM + lane * 8 + j] = (xs[j] - mu) * rs;
    }
    __syncthreads();
  }

  // store phase: thread t owns column t across 16 rows
#pragma unroll
  for (int r = 0; r < 16; r++) {
    float x = Cs[r * DDIM + t];
    if constexpr (ST_F32)  outF[(size_t)(row0 + r) * DDIM + t] = x;
    if constexpr (ST_F16)  outH[(size_t)(row0 + r) * DDIM + t] = (_Float16)x;
    if constexpr (ST_F16T) outHT[(size_t)t * Mtot + row0 + r]  = (_Float16)x;
  }
}

// ---------------------------------------------------------------------------
// Scores: S[N,N] = Qh @ Kh^T.  Kh (row-major [n][k]) is directly B^T.
// Block computes 64x64 (8 waves x 2 tiles). Raw scores to attn region.
// ---------------------------------------------------------------------------
__global__ void scores_kernel(const _Float16* __restrict__ Qh,
                              const _Float16* __restrict__ Kh,
                              float* __restrict__ S) {
  const int row0 = blockIdx.y * 64;
  const int col0 = blockIdx.x * 64;
  const int wave = threadIdx.x >> 5;
  const int lane = threadIdx.x & 31;
  const int mt = wave & 3;
  const int nt = (wave >> 2) * 2;

  const _Float16* arow = Qh + (size_t)(row0 + mt * 16) * DDIM;
  const _Float16* b0r  = Kh + (size_t)(col0 + nt * 16) * DDIM;
  const _Float16* b1r  = b0r + 16 * DDIM;

  v8f acc0 = {}, acc1 = {};
#pragma unroll
  for (int k0 = 0; k0 < DDIM; k0 += 32) {
    v16h a  = load_fragA(arow + k0, DDIM, lane);
    v16h b0 = load_fragB(b0r + k0, DDIM, lane);
    v16h b1 = load_fragB(b1r + k0, DDIM, lane);
    acc0 = wmma_f16(a, b0, acc0);
    acc1 = wmma_f16(a, b1, acc1);
  }
  int colw = lane & 15;
  int rh = (lane >> 4) * 8;
#pragma unroll
  for (int e = 0; e < 8; e++) {
    size_t rr = (size_t)(row0 + mt * 16 + e + rh) * N_NODES;
    S[rr + col0 + nt * 16 + colw]        = acc0[e];
    S[rr + col0 + (nt + 1) * 16 + colw]  = acc1[e];
  }
}

// ---------------------------------------------------------------------------
// Softmax over each row of S (length 8192), register-resident (32 vals/thread)
// ---------------------------------------------------------------------------
__global__ void softmax_kernel(float* __restrict__ S) {
  const int row = blockIdx.x;
  const int t = threadIdx.x;
  float* p = S + (size_t)row * N_NODES + t;
  __shared__ float red[8];

  float v[32];
  float mx = -3.0e38f;
#pragma unroll
  for (int j = 0; j < 32; j++) {
    v[j] = p[j * 256];
    mx = fmaxf(mx, v[j]);
  }
#pragma unroll
  for (int off = 16; off; off >>= 1) mx = fmaxf(mx, __shfl_xor(mx, off, 32));
  if ((t & 31) == 0) red[t >> 5] = mx;
  __syncthreads();
  mx = red[0];
#pragma unroll
  for (int i = 1; i < 8; i++) mx = fmaxf(mx, red[i]);
  __syncthreads();

  float s = 0.f;
#pragma unroll
  for (int j = 0; j < 32; j++) {
    v[j] = __expf(v[j] - mx);
    s += v[j];
  }
#pragma unroll
  for (int off = 16; off; off >>= 1) s += __shfl_xor(s, off, 32);
  if ((t & 31) == 0) red[t >> 5] = s;
  __syncthreads();
  s = red[0];
#pragma unroll
  for (int i = 1; i < 8; i++) s += red[i];
  float inv = 1.0f / s;
#pragma unroll
  for (int j = 0; j < 32; j++) p[j * 256] = v[j] * inv;
}

// ---------------------------------------------------------------------------
// Z_A = attn(f32 -> f16 via wide LDS stage) @ Vt^T.  Vt is f16 [256][8192].
// Block: 16 rows x 256 cols. A staged 16x128 per barrier pair (4 WMMA steps),
// with global_prefetch of the next attn tile.
// ---------------------------------------------------------------------------
__global__ void av_kernel(const float* __restrict__ S,
                          const _Float16* __restrict__ Vt,
                          float* __restrict__ ZA) {
  __shared__ _Float16 As[16 * 128];   // 4 KB
  const int row0 = blockIdx.x * 16;
  const int t = threadIdx.x;
  const int wave = t >> 5;
  const int lane = t & 31;
  const int n0 = wave * 32;

  v8f acc0 = {}, acc1 = {};
  const int sr = t >> 4;        // 0..15 : row of A tile
  const int sc = (t & 15) * 8;  // 0..120: 8 consecutive k per thread
  const float* srcbase = S + (size_t)(row0 + sr) * N_NODES + sc;
  const _Float16* bt0 = Vt + (size_t)n0 * N_NODES;
  const _Float16* bt1 = Vt + (size_t)(n0 + 16) * N_NODES;

  for (int k0 = 0; k0 < N_NODES; k0 += 128) {
    float4 x0 = *reinterpret_cast<const float4*>(srcbase + k0);
    float4 x1 = *reinterpret_cast<const float4*>(srcbase + k0 + 4);
    if (k0 + 128 < N_NODES)
      __builtin_prefetch(srcbase + k0 + 128, 0, 1);   // global_prefetch_b8
    union { uint4 u; _Float16 h[8]; } pk;
    pk.h[0] = (_Float16)x0.x; pk.h[1] = (_Float16)x0.y;
    pk.h[2] = (_Float16)x0.z; pk.h[3] = (_Float16)x0.w;
    pk.h[4] = (_Float16)x1.x; pk.h[5] = (_Float16)x1.y;
    pk.h[6] = (_Float16)x1.z; pk.h[7] = (_Float16)x1.w;
    *reinterpret_cast<uint4*>(&As[sr * 128 + sc]) = pk.u;   // one ds_store_b128
    __syncthreads();
#pragma unroll
    for (int kk = 0; kk < 128; kk += 32) {
      v16h a  = load_fragA(As + kk, 128, lane);
      v16h b0 = load_fragB(bt0 + k0 + kk, N_NODES, lane);
      v16h b1 = load_fragB(bt1 + k0 + kk, N_NODES, lane);
      acc0 = wmma_f16(a, b0, acc0);
      acc1 = wmma_f16(a, b1, acc1);
    }
    __syncthreads();
  }
  int colw = lane & 15;
  int rh = (lane >> 4) * 8;
#pragma unroll
  for (int e = 0; e < 8; e++) {
    ZA[(size_t)(row0 + e + rh) * DDIM + n0 + colw]      = acc0[e];
    ZA[(size_t)(row0 + e + rh) * DDIM + n0 + 16 + colw] = acc1[e];
  }
}

// ---------------------------------------------------------------------------
// GNN: segment_sum over sorted edge_row. Block = node, thread = feature dim.
// ---------------------------------------------------------------------------
__global__ void gnn_kernel(const float* __restrict__ H,
                           const int* __restrict__ erow,
                           const int* __restrict__ ecol,
                           const float* __restrict__ ew,
                           float* __restrict__ out, int E) {
  const int n = blockIdx.x;
  const int t = threadIdx.x;
  int lo = 0, hi = E;
  while (lo < hi) { int m = (lo + hi) >> 1; if (erow[m] < n) lo = m + 1; else hi = m; }
  int beg = lo;
  hi = E;
  while (lo < hi) { int m = (lo + hi) >> 1; if (erow[m] < n + 1) lo = m + 1; else hi = m; }
  int end = lo;

  float acc = 0.f;
  for (int e = beg; e < end; e++) {
    acc += H[(size_t)ecol[e] * DDIM + t] * ew[e];
  }
  out[(size_t)n * DDIM + t] = acc;
}

// ---------------------------------------------------------------------------
// Y = (0.5*Z_A + 0.5*gnn) @ W_cls + b_cls
// ---------------------------------------------------------------------------
__global__ void final_kernel(const float* __restrict__ ZA,
                             const float* __restrict__ G,
                             const float* __restrict__ Wc,
                             const float* __restrict__ bc,
                             float* __restrict__ Y) {
  int idx = blockIdx.x * blockDim.x + threadIdx.x;
  if (idx >= N_NODES * NCLS) return;
  int n = idx / NCLS, c = idx % NCLS;
  const float* za = ZA + (size_t)n * DDIM;
  const float* g  = G  + (size_t)n * DDIM;
  float acc = bc[c];
#pragma unroll 4
  for (int k = 0; k < DDIM; k++)
    acc += (0.5f * za[k] + 0.5f * g[k]) * Wc[k * NCLS + c];
  Y[idx] = acc;
}

// ---------------------------------------------------------------------------
extern "C" void kernel_launch(void* const* d_in, const int* in_sizes, int n_in,
                              void* d_out, int out_size, void* d_ws, size_t ws_size,
                              hipStream_t stream) {
  const float* X      = (const float*)d_in[0];
  const int*   erow   = (const int*)d_in[1];
  const int*   ecol   = (const int*)d_in[2];
  const float* ew     = (const float*)d_in[3];
  const float* W_emb  = (const float*)d_in[4];
  const float* b_emb  = (const float*)d_in[5];
  const float* W_q    = (const float*)d_in[6];
  const float* b_q    = (const float*)d_in[7];
  const float* W_k    = (const float*)d_in[8];
  const float* b_k    = (const float*)d_in[9];
  const float* W_v    = (const float*)d_in[10];
  const float* b_v    = (const float*)d_in[11];
  const float* W_gnn  = (const float*)d_in[12];
  const float* b_gnn  = (const float*)d_in[13];
  const float* W_cls  = (const float*)d_in[14];
  const float* b_cls  = (const float*)d_in[15];
  const int E = in_sizes[1];

  // workspace carve
  char* w = (char*)d_ws;
  size_t off = 0;
  auto carve = [&](size_t bytes) { void* p = w + off; off += (bytes + 255) & ~(size_t)255; return p; };
  _Float16* Xh    = (_Float16*)carve((size_t)N_NODES * F_IN * 2);
  _Float16* WembT = (_Float16*)carve((size_t)DDIM * F_IN * 2);
  _Float16* WqT   = (_Float16*)carve((size_t)DDIM * DDIM * 2);
  _Float16* WkT   = (_Float16*)carve((size_t)DDIM * DDIM * 2);
  _Float16* WvT   = (_Float16*)carve((size_t)DDIM * DDIM * 2);
  _Float16* WgT   = (_Float16*)carve((size_t)DDIM * DDIM * 2);
  _Float16* Z0h   = (_Float16*)carve((size_t)N_NODES * DDIM * 2);
  _Float16* Qh    = (_Float16*)carve((size_t)N_NODES * DDIM * 2);
  _Float16* Kh    = (_Float16*)carve((size_t)N_NODES * DDIM * 2);
  _Float16* Vt    = (_Float16*)carve((size_t)DDIM * N_NODES * 2);   // [256][8192]
  float*    Hf    = (float*)carve((size_t)N_NODES * DDIM * 4);
  float*    ZA    = (float*)carve((size_t)N_NODES * DDIM * 4);
  float*    Gout  = (float*)carve((size_t)N_NODES * DDIM * 4);

  float* Yout = (float*)d_out;                       // [8192,40]
  float* attn = Yout + (size_t)N_NODES * NCLS;       // [8192,8192]

  // prep
  cvt_f32_to_f16_kernel<<<(N_NODES * F_IN + 255) / 256, 256, 0, stream>>>(X, Xh, N_NODES * F_IN);
  transpose_f32_to_f16_kernel<<<(F_IN * DDIM + 255) / 256, 256, 0, stream>>>(W_emb, WembT, F_IN, DDIM);
  transpose_f32_to_f16_kernel<<<(DDIM * DDIM + 255) / 256, 256, 0, stream>>>(W_q, WqT, DDIM, DDIM);
  transpose_f32_to_f16_kernel<<<(DDIM * DDIM + 255) / 256, 256, 0, stream>>>(W_k, WkT, DDIM, DDIM);
  transpose_f32_to_f16_kernel<<<(DDIM * DDIM + 255) / 256, 256, 0, stream>>>(W_v, WvT, DDIM, DDIM);
  transpose_f32_to_f16_kernel<<<(DDIM * DDIM + 255) / 256, 256, 0, stream>>>(W_gnn, WgT, DDIM, DDIM);

  const int gblocks = N_NODES / 16;  // 512
  // Z0 = X @ W_emb + b   -> f16
  gemm_row16_kernel<false, false, true, false><<<gblocks, 256, 0, stream>>>(
      Xh, F_IN, WembT, b_emb, nullptr, Z0h, nullptr, N_NODES);
  // Q = LN(Z0 @ W_q + b) -> f16
  gemm_row16_kernel<true, false, true, false><<<gblocks, 256, 0, stream>>>(
      Z0h, DDIM, WqT, b_q, nullptr, Qh, nullptr, N_NODES);
  // K = LN(Z0 @ W_k + b) -> f16
  gemm_row16_kernel<true, false, true, false><<<gblocks, 256, 0, stream>>>(
      Z0h, DDIM, WkT, b_k, nullptr, Kh, nullptr, N_NODES);
  // V = Z0 @ W_v + b     -> f16 transposed [256][8192]
  gemm_row16_kernel<false, false, false, true><<<gblocks, 256, 0, stream>>>(
      Z0h, DDIM, WvT, b_v, nullptr, nullptr, Vt, N_NODES);
  // H = Z0 @ W_gnn + b   -> f32
  gemm_row16_kernel<false, true, false, false><<<gblocks, 256, 0, stream>>>(
      Z0h, DDIM, WgT, b_gnn, Hf, nullptr, nullptr, N_NODES);

  // attention
  scores_kernel<<<dim3(N_NODES / 64, N_NODES / 64), 256, 0, stream>>>(Qh, Kh, attn);
  softmax_kernel<<<N_NODES, 256, 0, stream>>>(attn);
  av_kernel<<<gblocks, 256, 0, stream>>>(attn, Vt, ZA);

  // sparse branch
  gnn_kernel<<<N_NODES, 256, 0, stream>>>(Hf, erow, ecol, ew, Gout, E);

  // blend + classifier
  final_kernel<<<(N_NODES * NCLS + 255) / 256, 256, 0, stream>>>(ZA, Gout, W_cls, b_cls, Yout);
}
